// Pointnet2Backbone_85993835200763
// MI455X (gfx1250) — matrix operand order repaired and checked by reading, not compiled, not run on hardware
//
#include <hip/hip_runtime.h>
#include <hip/hip_bf16.h>
#include <stdint.h>

// ---------- types ----------
typedef __attribute__((ext_vector_type(16))) __bf16 v16bf;
typedef __attribute__((ext_vector_type(8)))  __bf16 v8bf;
typedef __attribute__((ext_vector_type(8)))  float  v8f;
typedef __attribute__((ext_vector_type(4)))  int    i32x4;

__device__ __forceinline__ __bf16 f2bf(float f) {
  union { float f; uint32_t u; } a; a.f = f;
  uint32_t r = a.u + 0x7FFFu + ((a.u >> 16) & 1u);   // round-to-nearest-even
  union { uint16_t s; __bf16 h; } o; o.s = (uint16_t)(r >> 16);
  return o.h;
}
__device__ __forceinline__ float bf2f(__bf16 h) {
  union { __bf16 h; uint16_t s; } i; i.h = h;
  union { uint32_t u; float f; } o; o.u = ((uint32_t)i.s) << 16;
  return o.f;
}

// ---------- FPS: one block per batch ----------
#define FPS_T 256
__global__ void fps_kernel(const float* __restrict__ xyz, int N, int npoint,
                           float* __restrict__ dist, int* __restrict__ inds) {
  const int b = blockIdx.x;
  const float* P = xyz + (size_t)b * N * 3;
  float* D = dist + (size_t)b * N;
  int*   I = inds + (size_t)b * npoint;
  __shared__ float sval[FPS_T];
  __shared__ int   sidx[FPS_T];
  __shared__ int   s_last;
  const int tid = threadIdx.x;
  for (int i = tid; i < N; i += FPS_T) D[i] = 1e10f;
  if (tid == 0) { I[0] = 0; s_last = 0; }
  __syncthreads();
  for (int s = 1; s < npoint; ++s) {
    const int last = s_last;
    const float px = P[last*3+0], py = P[last*3+1], pz = P[last*3+2];
    float bestv = -1.0f; int besti = 0;
    for (int i = tid; i < N; i += FPS_T) {
      const float dx = P[i*3+0]-px, dy = P[i*3+1]-py, dz = P[i*3+2]-pz;
      float d = dx*dx + dy*dy + dz*dz;
      float dd = D[i]; dd = dd < d ? dd : d; D[i] = dd;
      if (dd > bestv) { bestv = dd; besti = i; }
    }
    sval[tid] = bestv; sidx[tid] = besti;
    __syncthreads();
    for (int off = FPS_T/2; off > 0; off >>= 1) {
      if (tid < off) {
        if (sval[tid+off] > sval[tid] ||
            (sval[tid+off] == sval[tid] && sidx[tid+off] < sidx[tid])) {
          sval[tid] = sval[tid+off]; sidx[tid] = sidx[tid+off];
        }
      }
      __syncthreads();
    }
    if (tid == 0) { I[s] = sidx[0]; s_last = sidx[0]; }
    __syncthreads();
  }
}

// ---------- gather new_xyz ----------
__global__ void gather_xyz(const float* __restrict__ xyz, const int* __restrict__ inds,
                           float* __restrict__ out, int N, int M, int total) {
  int t = blockIdx.x * blockDim.x + threadIdx.x;
  if (t >= total) return;
  int c = t % 3; int bm = t / 3;
  int m = bm % M; int b = bm / M;
  int p = inds[(size_t)b * M + m];
  out[t] = xyz[((size_t)b * N + p) * 3 + c];
}

// ---------- ball query (reference semantics: ascending indices, pad first, N-1 fallback) ----------
__global__ void ball_query(const float* __restrict__ xyz, const float* __restrict__ new_xyz,
                           int N, int M, int nsample, float r2,
                           int* __restrict__ idx, int total) {
  int q = blockIdx.x * blockDim.x + threadIdx.x;
  if (q >= total) return;
  int b = q / M;
  const float* P = xyz + (size_t)b * N * 3;
  const float cx = new_xyz[q*3+0], cy = new_xyz[q*3+1], cz = new_xyz[q*3+2];
  int* out = idx + (size_t)q * nsample;
  int cnt = 0, first = N - 1;
  for (int j = 0; j < N && cnt < nsample; ++j) {
    const float dx = P[j*3+0]-cx, dy = P[j*3+1]-cy, dz = P[j*3+2]-cz;
    if (dx*dx + dy*dy + dz*dz < r2) {
      if (cnt == 0) first = j;
      out[cnt++] = j;
    }
  }
  for (; cnt < nsample; ++cnt) out[cnt] = first;
}

// ---------- build grouped A matrix (bf16, K padded to mult of 32) ----------
__global__ void build_sa_A(const float* __restrict__ xyz, const float* __restrict__ new_xyz,
                           const float* __restrict__ feat, const int* __restrict__ idx,
                           int N, int M, int nsample, int Cf, int Kpad, float inv_r,
                           __bf16* __restrict__ A, unsigned long long total) {
  unsigned long long t = (unsigned long long)blockIdx.x * blockDim.x + threadIdx.x;
  if (t >= total) return;
  int k = (int)(t % Kpad);
  unsigned long long row = t / Kpad;
  unsigned long long bm = row / nsample;
  int m = (int)(bm % M); int b = (int)(bm / M);
  int p = idx[row];
  float v;
  if (k < 3)            v = (xyz[((size_t)b*N + p)*3 + k] - new_xyz[((size_t)b*M + m)*3 + k]) * inv_r;
  else if (k < 3 + Cf)  v = feat[((size_t)b*N + p)*Cf + (k - 3)];
  else                  v = 0.0f;
  A[t] = f2bf(v);
}

// ---------- weight convert f32(N,K) -> bf16(N,Kpad), zero padded ----------
__global__ void convert_w(const float* __restrict__ W, __bf16* __restrict__ Wb,
                          int K, int Kpad, int total) {
  int t = blockIdx.x * blockDim.x + threadIdx.x;
  if (t >= total) return;
  int k = t % Kpad; int n = t / Kpad;
  Wb[t] = f2bf(k < K ? W[(size_t)n*K + k] : 0.0f);
}

// ---------- WMMA GEMM: D = relu(A[M,Kpad] * Wb[N,Kpad]^T + bias) ----------
// Block = 256 threads (8 waves) -> 256x64 output block.
// B panel (64 out-channels x Kpad) staged in LDS via async load-to-LDS (ASYNCcnt),
// shared by all 8 waves. Each wave: 32 rows x 64 cols = 2 A fragments x 4 B
// fragments -> 8 independent WMMAs per K-step; B fragments preloaded into
// distinct registers so WMMAs don't serialize on ds waits.
template <bool OUT_BF16>
__global__ void wmma_gemm_bias_relu(const __bf16* __restrict__ A, int lda,
                                    const __bf16* __restrict__ Wb,
                                    const float* __restrict__ bias,
                                    void* __restrict__ Out, int M, int N) {
  __shared__ __bf16 sB[64 * 512];          // 64 rows x (Kpad<=512) bf16 = 64KB max
  const int tid  = threadIdx.x;
  const int lane = tid & 31;
  const int wave = tid >> 5;
  const int colBlocks = N >> 6;            // N / 64
  const int cb = blockIdx.x % colBlocks;
  const int rb = blockIdx.x / colBlocks;
  const int n_base = cb << 6;

  // ---- stage B panel [64 x lda] into LDS ----
  const __bf16* gB = Wb + (size_t)n_base * lda;
  const int chunks = (64 * lda) >> 3;      // 16-byte chunks
#if defined(__AMDGCN__) && __has_builtin(__builtin_amdgcn_global_load_async_to_lds_b128)
  for (int c = tid; c < chunks; c += 256) {
    __builtin_amdgcn_global_load_async_to_lds_b128(
        (__attribute__((address_space(1))) i32x4*)(gB + (size_t)c * 8),
        (__attribute__((address_space(3))) i32x4*)(sB + (size_t)c * 8),
        0, 0);
  }
  asm volatile("s_wait_asynccnt 0x0" ::: "memory");
#else
  for (int c = tid; c < chunks; c += 256) {
    *(v8bf*)(sB + (size_t)c * 8) = *(const v8bf*)(gB + (size_t)c * 8);
  }
#endif
  __syncthreads();

  // ---- compute: wave -> 32 rows (2 A frags), 4 column tiles ----
  const int  m0  = rb * 256 + wave * 32;
  const bool hi  = lane >= 16;
  const int  l16 = lane & 15;
  const __bf16* Arow0 = A + (size_t)(m0 + l16) * lda;
  const __bf16* Arow1 = A + (size_t)(m0 + 16 + l16) * lda;
  const __bf16* sRow  = sB + (size_t)l16 * lda;
  union AB { v16bf v; v8bf h[2]; };
  v8f acc[8] = {};
  for (int kb = 0; kb < lda; kb += 32) {
    // A 16x32 bf16 layout: lanes 0-15: K {0..7,16..23}; lanes 16-31: K {8..15,24..31}
    AB a0, a1;
    a0.h[0] = *(const v8bf*)(Arow0 + kb + (hi ? 8  : 0));
    a0.h[1] = *(const v8bf*)(Arow0 + kb + (hi ? 24 : 16));
    a1.h[0] = *(const v8bf*)(Arow1 + kb + (hi ? 8  : 0));
    a1.h[1] = *(const v8bf*)(Arow1 + kb + (hi ? 24 : 16));
    // B 32x16 bf16 layout: lanes 0-15: K {0..7,16..23}; lanes 16-31: K {8..15,24..31}
    const int kb2 = kb + (hi ? 8 : 0);
    AB bf[4];
#pragma unroll
    for (int t = 0; t < 4; ++t) {
      const __bf16* br = sRow + (size_t)(t << 4) * lda + kb2;
      bf[t].h[0] = *(const v8bf*)br;
      bf[t].h[1] = *(const v8bf*)(br + 16);
    }
#pragma unroll
    for (int t = 0; t < 4; ++t)
      acc[t] = __builtin_amdgcn_wmma_f32_16x16x32_bf16(false, a0.v, false, bf[t].v,
                                                       (short)0, acc[t], false, false);
#pragma unroll
    for (int t = 0; t < 4; ++t)
      acc[4 + t] = __builtin_amdgcn_wmma_f32_16x16x32_bf16(false, a1.v, false, bf[t].v,
                                                           (short)0, acc[4 + t], false, false);
  }

  // ---- epilogue: bias + relu + store (C/D layout: lane -> N, VGPR r -> M) ----
#pragma unroll
  for (int g = 0; g < 2; ++g) {
    const int mg = m0 + g * 16;
#pragma unroll
    for (int t = 0; t < 4; ++t) {
      const int n = n_base + (t << 4) + l16;
      const float bn = bias[n];
#pragma unroll
      for (int r = 0; r < 8; ++r) {
        const int m = mg + r + (hi ? 8 : 0);
        float v = acc[g * 4 + t][r] + bn;
        v = v > 0.0f ? v : 0.0f;
        if (OUT_BF16) ((__bf16*)Out)[(size_t)m * N + n] = f2bf(v);
        else          ((float*) Out)[(size_t)m * N + n] = v;
      }
    }
  }
}

// ---------- max over nsample ----------
__global__ void maxpool_kernel(const __bf16* __restrict__ act, int nsample, int C,
                               float* __restrict__ out, unsigned long long total) {
  unsigned long long t = (unsigned long long)blockIdx.x * blockDim.x + threadIdx.x;
  if (t >= total) return;
  int c = (int)(t % C);
  unsigned long long bm = t / C;
  const __bf16* p = act + (bm * nsample) * C + c;
  float mx = -1e30f;
  for (int s = 0; s < nsample; ++s) {
    float v = bf2f(p[(size_t)s * C]);
    mx = v > mx ? v : mx;
  }
  out[t] = mx;
}

// ---------- 3-NN + inverse-distance weights ----------
__global__ void knn3_kernel(const float* __restrict__ xyzA, const float* __restrict__ xyzB,
                            int M1, int M2, int* __restrict__ idx3, float* __restrict__ w3,
                            int total) {
  int q = blockIdx.x * blockDim.x + threadIdx.x;
  if (q >= total) return;
  int b = q / M1;
  const float cx = xyzA[q*3+0], cy = xyzA[q*3+1], cz = xyzA[q*3+2];
  const float* P = xyzB + (size_t)b * M2 * 3;
  float d0 = 1e30f, d1 = 1e30f, d2 = 1e30f;
  int i0 = 0, i1 = 0, i2 = 0;
  for (int j = 0; j < M2; ++j) {
    const float dx = P[j*3+0]-cx, dy = P[j*3+1]-cy, dz = P[j*3+2]-cz;
    const float d = dx*dx + dy*dy + dz*dz;
    if (d < d0)      { d2 = d1; i2 = i1; d1 = d0; i1 = i0; d0 = d; i0 = j; }
    else if (d < d1) { d2 = d1; i2 = i1; d1 = d;  i1 = j; }
    else if (d < d2) { d2 = d;  i2 = j; }
  }
  float w0 = 1.0f / (d0 + 1e-8f), w1 = 1.0f / (d1 + 1e-8f), w2 = 1.0f / (d2 + 1e-8f);
  const float inv = 1.0f / (w0 + w1 + w2);
  idx3[q*3+0] = i0; idx3[q*3+1] = i1; idx3[q*3+2] = i2;
  w3[q*3+0] = w0*inv; w3[q*3+1] = w1*inv; w3[q*3+2] = w2*inv;
}

// ---------- build FP concat row: [interp(C) | feat1(C)] in bf16 ----------
__global__ void build_fp_A(const float* __restrict__ feat1, const float* __restrict__ feat2,
                           const int* __restrict__ idx3, const float* __restrict__ w3,
                           int M1, int M2, int C, __bf16* __restrict__ A,
                           unsigned long long total) {
  unsigned long long t = (unsigned long long)blockIdx.x * blockDim.x + threadIdx.x;
  if (t >= total) return;
  const int twoC = 2 * C;
  int k = (int)(t % twoC);
  unsigned long long row = t / twoC;      // row = b*M1 + i
  int b = (int)(row / M1);
  float v;
  if (k < C) {
    v = 0.0f;
#pragma unroll
    for (int j = 0; j < 3; ++j) {
      const int p = idx3[row*3 + j];
      v += w3[row*3 + j] * feat2[((size_t)b * M2 + p) * C + k];
    }
  } else {
    v = feat1[row * C + (k - C)];
  }
  A[t] = f2bf(v);
}

// ---------- host orchestration ----------
extern "C" void kernel_launch(void* const* d_in, const int* in_sizes, int n_in,
                              void* d_out, int out_size, void* d_ws, size_t ws_size,
                              hipStream_t stream) {
  (void)in_sizes; (void)n_in; (void)out_size; (void)ws_size;
  const int B = 2, N0 = 16384;
  const float* pc = (const float*)d_in[0];

  // bump allocator over workspace
  char* base = (char*)d_ws; size_t off = 0;
  auto alloc = [&](size_t bytes) -> void* {
    off = (off + 255) & ~(size_t)255;
    void* p = base + off; off += bytes; return p;
  };

  float* dist = (float*)alloc((size_t)B * N0 * 4);
  int*   inds = (int*)  alloc((size_t)B * 2048 * 4);
  int*   idxb = (int*)  alloc((size_t)B * 2048 * 64 * 4);
  float* xyz1 = (float*)alloc((size_t)B * 2048 * 3 * 4);
  float* xyz2 = (float*)alloc((size_t)B * 1024 * 3 * 4);
  float* xyz3 = (float*)alloc((size_t)B * 512  * 3 * 4);
  float* xyz4 = (float*)alloc((size_t)B * 256  * 3 * 4);
  float* f1   = (float*)alloc((size_t)B * 2048 * 128 * 4);
  float* f2   = (float*)alloc((size_t)B * 1024 * 256 * 4);
  float* f3   = (float*)alloc((size_t)B * 512  * 256 * 4);
  float* f4   = (float*)alloc((size_t)B * 256  * 256 * 4);
  float* ffp1 = (float*)alloc((size_t)B * 512  * 256 * 4);
  int*   idx3 = (int*)  alloc((size_t)B * 1024 * 3 * 4);
  float* w3   = (float*)alloc((size_t)B * 1024 * 3 * 4);
  __bf16* Wb  = (__bf16*)alloc((size_t)256 * 512 * 2);            // max N*Kpad
  __bf16* fpA = (__bf16*)alloc((size_t)B * 1024 * 512 * 2);
  __bf16* fpH = (__bf16*)alloc((size_t)B * 1024 * 256 * 2);
  __bf16* Abuf= (__bf16*)alloc((size_t)B * 1024 * 32 * 160 * 2);  // max rows*Kpad (SA2)
  __bf16* H1  = (__bf16*)alloc((size_t)262144 * 128 * 2);
  __bf16* H2  = (__bf16*)alloc((size_t)262144 * 64  * 2);

  auto gemm = [&](const __bf16* A, int Kpad, const float* W, int Korig,
                  const float* bias, int Ncols, void* out, bool out_bf16, int M) {
    const int wtot = Ncols * Kpad;
    convert_w<<<(wtot + 255) / 256, 256, 0, stream>>>(W, Wb, Korig, Kpad, wtot);
    const int blocks = (M / 256) * (Ncols / 64);   // 256x64 output per block
    if (out_bf16)
      wmma_gemm_bias_relu<true ><<<blocks, 256, 0, stream>>>(A, Kpad, Wb, bias, out, M, Ncols);
    else
      wmma_gemm_bias_relu<false><<<blocks, 256, 0, stream>>>(A, Kpad, Wb, bias, out, M, Ncols);
  };

  auto run_sa = [&](const float* xyz_in, int N, const float* feat_in, int Cf,
                    int npoint, int nsample, float radius, int widx,
                    int n0c, int n1c, int n2c, float* new_xyz, float* f_out) {
    fps_kernel<<<B, FPS_T, 0, stream>>>(xyz_in, N, npoint, dist, inds);
    const int tg = B * npoint * 3;
    gather_xyz<<<(tg + 255) / 256, 256, 0, stream>>>(xyz_in, inds, new_xyz, N, npoint, tg);
    const int tq = B * npoint;
    ball_query<<<(tq + 255) / 256, 256, 0, stream>>>(xyz_in, new_xyz, N, npoint, nsample,
                                                     radius * radius, idxb, tq);
    const int K0 = 3 + Cf, Kpad0 = (K0 + 31) & ~31;
    const unsigned long long Mrows = (unsigned long long)B * npoint * nsample;
    const unsigned long long ta = Mrows * Kpad0;
    build_sa_A<<<(unsigned)((ta + 255) / 256), 256, 0, stream>>>(
        xyz_in, new_xyz, feat_in, idxb, N, npoint, nsample, Cf, Kpad0, 1.0f / radius, Abuf, ta);
    const float* W0 = (const float*)d_in[widx + 0]; const float* b0 = (const float*)d_in[widx + 1];
    const float* W1 = (const float*)d_in[widx + 2]; const float* b1 = (const float*)d_in[widx + 3];
    const float* W2 = (const float*)d_in[widx + 4]; const float* b2 = (const float*)d_in[widx + 5];
    gemm(Abuf, Kpad0, W0, K0,  b0, n0c, H1, true, (int)Mrows);
    gemm(H1,   n0c,   W1, n0c, b1, n1c, H2, true, (int)Mrows);
    gemm(H2,   n1c,   W2, n1c, b2, n2c, H1, true, (int)Mrows);
    const unsigned long long tp = (unsigned long long)B * npoint * n2c;
    maxpool_kernel<<<(unsigned)((tp + 255) / 256), 256, 0, stream>>>(H1, nsample, n2c, f_out, tp);
  };

  auto run_fp = [&](const float* xyzHi, int M1, const float* xyzLo, int M2,
                    const float* featHi, const float* featLo, int C,
                    int widx, float* out_f) {
    const int tq = B * M1;
    knn3_kernel<<<(tq + 255) / 256, 256, 0, stream>>>(xyzHi, xyzLo, M1, M2, idx3, w3, tq);
    const unsigned long long ta = (unsigned long long)B * M1 * 2 * C;
    build_fp_A<<<(unsigned)((ta + 255) / 256), 256, 0, stream>>>(
        featHi, featLo, idx3, w3, M1, M2, C, fpA, ta);
    const float* W0 = (const float*)d_in[widx + 0]; const float* b0 = (const float*)d_in[widx + 1];
    const float* W1 = (const float*)d_in[widx + 2]; const float* b1 = (const float*)d_in[widx + 3];
    const int M = B * M1;
    gemm(fpA, 2 * C, W0, 2 * C, b0, 256, fpH, true, M);
    gemm(fpH, 256,   W1, 256,   b1, 256, out_f, false, M);
  };

  // ---- pipeline ----
  run_sa(pc,   N0,   nullptr, 0,   2048, 64, 0.2f,  1, 64, 64, 128,  xyz1, f1);
  run_sa(xyz1, 2048, f1,      128, 1024, 32, 0.4f,  7, 128, 128, 256, xyz2, f2);
  run_sa(xyz2, 1024, f2,      256, 512,  16, 0.8f, 13, 128, 128, 256, xyz3, f3);
  run_sa(xyz3, 512,  f3,      256, 256,  16, 1.2f, 19, 128, 128, 256, xyz4, f4);
  run_fp(xyz3, 512,  xyz4, 256, f3, f4,   256, 25, ffp1);
  run_fp(xyz2, 1024, xyz3, 512, f2, ffp1, 256, 29, (float*)d_out + (size_t)B * 1024 * 3);
  (void)hipMemcpyAsync(d_out, xyz2, (size_t)B * 1024 * 3 * 4, hipMemcpyDeviceToDevice, stream);
}
